// Cell_dis_Auto_51290499449118
// MI455X (gfx1250) — compile-verified, matrix operand order checked by reading, not run on hardware
//
#include <hip/hip_runtime.h>
#include <hip/hip_bf16.h>
#include <math.h>

typedef __bf16 bf16;
typedef __attribute__((ext_vector_type(16))) __bf16 v16bf;
typedef __attribute__((ext_vector_type(8)))  __bf16 v8bf;
typedef __attribute__((ext_vector_type(8)))  float  v8f;

#define BATCH 32
#define CFULL 256
#define CH    64
#define HW    4096
#define WIDTH 64
#define AC    384          // 6*CH
#define OPST  8388608ULL   // 32*64*4096 floats per op slab

// ---------------------------------------------------------------- relu + per-(b,c) mean/max
__global__ __launch_bounds__(256) void k_relu_stats(const float* __restrict__ src,
                                                    float* __restrict__ dst,
                                                    float* __restrict__ omean,
                                                    float* __restrict__ omax) {
  int c = blockIdx.x, b = blockIdx.y;
  const float* p = src + ((size_t)(b * CFULL + c)) * HW;
  float* q = dst + ((size_t)(b * CFULL + c)) * HW;
  float s = 0.f, m = -INFINITY;
  for (int i = threadIdx.x; i < HW; i += 256) {
    float v = p[i];
    v = v > 0.f ? v : 0.f;
    q[i] = v;
    s += v;
    m = fmaxf(m, v);
  }
  __shared__ float ss[256], sm[256];
  ss[threadIdx.x] = s; sm[threadIdx.x] = m;
  __syncthreads();
  for (int o = 128; o > 0; o >>= 1) {
    if (threadIdx.x < o) {
      ss[threadIdx.x] += ss[threadIdx.x + o];
      sm[threadIdx.x] = fmaxf(sm[threadIdx.x], sm[threadIdx.x + o]);
    }
    __syncthreads();
  }
  if (threadIdx.x == 0) {
    omean[b * CFULL + c] = ss[0] * (1.f / HW);
    omax[b * CFULL + c]  = sm[0];
  }
}

// ---------------------------------------------------------------- channel attention MLP -> nl
__global__ __launch_bounds__(256) void k_ca_mlp(const float* __restrict__ mean,
                                                const float* __restrict__ mx,
                                                const float* __restrict__ w1,   // [128][256]
                                                const float* __restrict__ w2,   // [256][128]
                                                float* __restrict__ nl) {
  int b = blockIdx.x;
  __shared__ float vm[256], vx[256], hm[128], hx[128];
  vm[threadIdx.x] = mean[b * CFULL + threadIdx.x];
  vx[threadIdx.x] = mx[b * CFULL + threadIdx.x];
  __syncthreads();
  int j = threadIdx.x & 127;
  const float* src = (threadIdx.x < 128) ? vm : vx;
  float s = 0.f;
  for (int k = 0; k < 256; ++k) s += w1[j * 256 + k] * src[k];
  s = s > 0.f ? s : 0.f;
  if (threadIdx.x < 128) hm[j] = s; else hx[j] = s;
  __syncthreads();
  float o = 0.f;
  for (int k = 0; k < 128; ++k) o += w2[threadIdx.x * 128 + k] * (hm[k] + hx[k]);
  nl[b * CFULL + threadIdx.x] = 1.f / (1.f + __expf(-o));
}

// ---------------------------------------------------------------- slist + top-64 (lax.top_k tie rules)
__global__ __launch_bounds__(256) void k_topk(const float* __restrict__ nl, int* __restrict__ idx) {
  __shared__ float sl[256];
  float s = 0.f;
  for (int b = 0; b < BATCH; ++b) s += nl[b * CFULL + threadIdx.x];
  sl[threadIdx.x] = s;
  __syncthreads();
  if (threadIdx.x == 0) {
    for (int k = 0; k < CH; ++k) {
      int bi = 0; float bv = sl[0];
      for (int c = 1; c < 256; ++c)
        if (sl[c] > bv) { bv = sl[c]; bi = c; }
      idx[k] = bi;
      sl[bi] = -INFINITY;
    }
  }
}

// ---------------------------------------------------------------- h *= nl (in place)
__global__ __launch_bounds__(256) void k_scale(float* __restrict__ h, const float* __restrict__ nl) {
  int c = blockIdx.x, b = blockIdx.y;
  float s = nl[b * CFULL + c];
  float* p = h + ((size_t)(b * CFULL + c)) * HW;
  for (int i = threadIdx.x; i < HW; i += 256) p[i] *= s;
}

// ---------------------------------------------------------------- gather xt: fp32 NCHW + bf16 NHWC (LDS transpose)
__global__ __launch_bounds__(256) void k_gather(const float* __restrict__ h,
                                                const int* __restrict__ idx,
                                                float* __restrict__ t5,
                                                bf16* __restrict__ xbf) {
  int b = blockIdx.y;
  int p0 = blockIdx.x << 6;           // 64-pixel tile
  __shared__ bf16 tile[64 * 65];      // [pixel][channel], padded stride
  __shared__ int sidx[64];
  if (threadIdx.x < 64) sidx[threadIdx.x] = idx[threadIdx.x];
  __syncthreads();
  for (int e = threadIdx.x; e < 4096; e += 256) {
    int j = e >> 6, p = e & 63;       // coalesced along p
    float v = h[((size_t)(b * CFULL + sidx[j])) * HW + p0 + p];
    t5[((size_t)(b * CH + j)) * HW + p0 + p] = v;
    tile[p * 65 + j] = (bf16)v;
  }
  __syncthreads();
  for (int e = threadIdx.x; e < 4096; e += 256) {
    int p = e >> 6, c = e & 63;       // coalesced along c (contiguous 128B per pixel)
    xbf[((size_t)(b * HW + p0 + p) << 6) + c] = tile[p * 65 + c];
  }
}

// ---------------------------------------------------------------- conv weight [co][cin][t] -> bf16 [t][co][cin]
__global__ __launch_bounds__(256) void k_wcvt(const float* __restrict__ w, bf16* __restrict__ o, int KK) {
  int i = blockIdx.x * 256 + threadIdx.x;
  int total = CH * CH * KK;
  if (i >= total) return;
  int t = i % KK;
  int cin = (i / KK) % CH;
  int co = i / (KK * CH);
  o[(t * CH + co) * CH + cin] = (bf16)w[i];
}

// ---------------------------------------------------------------- bf16 fragment load (K-split per ISA layout)
__device__ inline v16bf load_frag(const bf16* p) {
  v8bf lo = *(const v8bf*)p;        // K = khalf .. khalf+7
  v8bf hi = *(const v8bf*)(p + 16); // K = 16+khalf .. 16+khalf+7
  v16bf r;
#pragma unroll
  for (int i = 0; i < 8; ++i) { r[i] = lo[i]; r[8 + i] = hi[i]; }
  return r;
}

// ---------------------------------------------------------------- implicit-GEMM conv via v_wmma_f32_16x16x32_bf16
// Input strip staged in LDS once per block (zero-padded halo); all taps + all
// 4 Cout-tile waves feed from LDS -> 4x less global B traffic, branch-free MMA loop.
template <int KS>
__global__ __launch_bounds__(256) void k_conv(const bf16* __restrict__ xbf,   // [B][H][W][64] bf16
                                              const bf16* __restrict__ wbf,   // [KS*KS][64][64] bf16
                                              float* __restrict__ out) {      // [B][64][4096] fp32
  constexpr int PAD = KS / 2;
  constexpr int WT = 32 + 2 * PAD;
  __shared__ bf16 tile[KS * WT * 64];
  int lane = threadIdx.x & 31, wv = threadIdx.x >> 5;
  int b = blockIdx.y;
  int base = blockIdx.x << 5;         // 32-pixel strip, single row
  int h0 = base >> 6, c0 = base & 63;

  // cooperative stage: [row][col][cin], 16B chunks, zero-fill out-of-bounds
  for (int e = threadIdx.x; e < KS * WT * 8; e += 256) {
    int ci8 = (e & 7) << 3;
    int col = (e >> 3) % WT;
    int row = (e >> 3) / WT;
    int gh = h0 + row - PAD;
    int gw = c0 + col - PAD;
    v8bf v;
    if (((unsigned)gh < 64u) && ((unsigned)gw < 64u)) {
      v = *(const v8bf*)(xbf + (((size_t)b * HW + (size_t)(gh * WIDTH + gw)) << 6) + ci8);
    } else {
#pragma unroll
      for (int i = 0; i < 8; ++i) v[i] = (bf16)0.f;
    }
    *(v8bf*)(tile + (row * WT + col) * 64 + ci8) = v;
  }
  __syncthreads();

  int m16 = lane & 15;
  int khalf = (lane >> 4) << 3;        // 0 or 8
  int co_base = (wv & 3) << 4;         // 0,16,32,48
  int nloc = ((wv >> 2) << 4) + m16;   // 0..31 within strip

  v8f acc = {};
#pragma unroll
  for (int t = 0; t < KS * KS; ++t) {
    int dy = t / KS, dx = t % KS;
    const bf16* ap = wbf + (((size_t)(t * CH + co_base + m16)) << 6) + khalf;
    const bf16* bp = tile + (dy * WT + nloc + dx) * 64 + khalf;
    __builtin_prefetch(ap, 0, 0);
#pragma unroll
    for (int kb = 0; kb < CH; kb += 32) {
      v16bf A = load_frag(ap + kb);
      v16bf Bm = load_frag(bp + kb);
      acc = __builtin_amdgcn_wmma_f32_16x16x32_bf16(false, A, false, Bm, (short)0, acc, false, false);
    }
  }
  int mrow = (lane >> 4) << 3;  // 0 or 8
  int pcol = base + nloc;
  float* q = out + ((size_t)(b * CH + co_base + mrow)) * HW + (size_t)pcol;
#pragma unroll
  for (int r = 0; r < 8; ++r) q[(size_t)r * HW] = acc[r];
}

// ---------------------------------------------------------------- 3x3 avg (valid-count) + max pool
__global__ __launch_bounds__(256) void k_pool(const float* __restrict__ xt,
                                              float* __restrict__ avg,
                                              float* __restrict__ mxp) {
  int c = blockIdx.x, b = blockIdx.y;
  const float* p = xt + ((size_t)(b * CH + c)) * HW;
  size_t o = ((size_t)(b * CH + c)) * HW;
  for (int i = threadIdx.x; i < HW; i += 256) {
    int h = i >> 6, w = i & 63;
    float s = 0.f, m = -INFINITY;
    int cnt = 0;
    for (int dy = -1; dy <= 1; ++dy) {
      int ih = h + dy;
      if ((unsigned)ih >= 64u) continue;
      for (int dx = -1; dx <= 1; ++dx) {
        int iw = w + dx;
        if ((unsigned)iw >= 64u) continue;
        float v = p[ih * WIDTH + iw];
        s += v; m = fmaxf(m, v); ++cnt;
      }
    }
    avg[o + i] = s / (float)cnt;
    mxp[o + i] = m;
  }
}

// ---------------------------------------------------------------- per-channel BN stats over (B,H,W)
__global__ __launch_bounds__(256) void k_bnstats(const float* __restrict__ t3,
                                                 const float* __restrict__ t4,
                                                 float* __restrict__ mv) {
  int c = blockIdx.x;
  const float* t = blockIdx.y ? t4 : t3;
  float s = 0.f, s2 = 0.f;
  for (int e = threadIdx.x; e < BATCH * HW; e += 256) {
    int b = e >> 12, i = e & 4095;
    float v = t[((size_t)(b * CH + c)) * HW + i];
    s += v; s2 += v * v;
  }
  __shared__ float a1[256], a2[256];
  a1[threadIdx.x] = s; a2[threadIdx.x] = s2;
  __syncthreads();
  for (int o = 128; o > 0; o >>= 1) {
    if (threadIdx.x < o) { a1[threadIdx.x] += a1[threadIdx.x + o]; a2[threadIdx.x] += a2[threadIdx.x + o]; }
    __syncthreads();
  }
  if (threadIdx.x == 0) {
    float m = a1[0] * (1.f / (BATCH * HW));
    float var = a2[0] * (1.f / (BATCH * HW)) - m * m;
    mv[(blockIdx.y * CH + c) * 2 + 0] = m;
    mv[(blockIdx.y * CH + c) * 2 + 1] = var;
  }
}

__global__ __launch_bounds__(256) void k_bnapply(float* __restrict__ t3, float* __restrict__ t4,
                                                 const float* __restrict__ mv) {
  int c = blockIdx.x, b = blockIdx.y;
  float* t = blockIdx.z ? t4 : t3;
  float m = mv[(blockIdx.z * CH + c) * 2 + 0];
  float r = rsqrtf(mv[(blockIdx.z * CH + c) * 2 + 1] + 1e-5f);
  float* p = t + ((size_t)(b * CH + c)) * HW;
  for (int i = threadIdx.x; i < HW; i += 256) p[i] = (p[i] - m) * r;
}

// ---------------------------------------------------------------- per-(b, op*64+c) spatial mean of temp
__global__ __launch_bounds__(256) void k_tmean(const float* __restrict__ temp, float* __restrict__ tm) {
  int oc = blockIdx.x, b = blockIdx.y;
  int op = oc >> 6, c = oc & 63;
  const float* p = temp + ((size_t)op * BATCH + b) * CH * HW + (size_t)c * HW;
  float s = 0.f;
  for (int i = threadIdx.x; i < HW; i += 256) s += p[i];
  __shared__ float ss[256];
  ss[threadIdx.x] = s;
  __syncthreads();
  for (int o = 128; o > 0; o >>= 1) {
    if (threadIdx.x < o) ss[threadIdx.x] += ss[threadIdx.x + o];
    __syncthreads();
  }
  if (threadIdx.x == 0) tm[b * AC + oc] = ss[0] * (1.f / HW);
}

// ---------------------------------------------------------------- op attention MLP -> y
__global__ __launch_bounds__(256) void k_att(const float* __restrict__ tm,
                                             const float* __restrict__ w1,  // [48][384]
                                             const float* __restrict__ w2,  // [384][48]
                                             float* __restrict__ y) {
  int b = blockIdx.x;
  __shared__ float v[AC], hid[48];
  for (int i = threadIdx.x; i < AC; i += 256) v[i] = tm[b * AC + i];
  __syncthreads();
  if (threadIdx.x < 48) {
    float s = 0.f;
    for (int k = 0; k < AC; ++k) s += w1[threadIdx.x * AC + k] * v[k];
    hid[threadIdx.x] = s > 0.f ? s : 0.f;
  }
  __syncthreads();
  for (int c = threadIdx.x; c < AC; c += 256) {
    float s = 0.f;
    for (int k = 0; k < 48; ++k) s += w2[c * 48 + k] * hid[k];
    y[b * AC + c] = 1.f / (1.f + __expf(-s));
  }
}

// ---------------------------------------------------------------- weighted op sum, scatter into h[:, idx]
__global__ __launch_bounds__(256) void k_combine(const float* __restrict__ temp,
                                                 const float* __restrict__ y,
                                                 const int* __restrict__ idx,
                                                 float* __restrict__ hbuf) {
  int c = blockIdx.x, b = blockIdx.y;
  float w0 = y[b * AC + 0 * CH + c], w1 = y[b * AC + 1 * CH + c], w2 = y[b * AC + 2 * CH + c];
  float w3 = y[b * AC + 3 * CH + c], w4 = y[b * AC + 4 * CH + c], w5 = y[b * AC + 5 * CH + c];
  int dc = idx[c];
  const float* p = temp + ((size_t)(b * CH + c)) * HW;
  float* q = hbuf + ((size_t)(b * CFULL + dc)) * HW;
  for (int i = threadIdx.x; i < HW; i += 256) {
    float s = p[i] * w0 + p[OPST + i] * w1 + p[2 * OPST + i] * w2 +
              p[3 * OPST + i] * w3 + p[4 * OPST + i] * w4 + p[5 * OPST + i] * w5;
    q[i] = s;
  }
}

// ---------------------------------------------------------------- out = h + x
__global__ __launch_bounds__(256) void k_add(const float* __restrict__ h, const float* __restrict__ x,
                                             float* __restrict__ out, size_t n) {
  size_t stride = (size_t)gridDim.x * 256;
  for (size_t i = (size_t)blockIdx.x * 256 + threadIdx.x; i < n; i += stride)
    out[i] = h[i] + x[i];
}

// ================================================================ host launcher
extern "C" void kernel_launch(void* const* d_in, const int* in_sizes, int n_in,
                              void* d_out, int out_size, void* d_ws, size_t ws_size,
                              hipStream_t stream) {
  const float* x = (const float*)d_in[0];
  struct SW { const float *caw1, *caw2, *w1, *w3, *w5, *aw1, *aw2; };
  SW st[2] = {
    {(const float*)d_in[2],  (const float*)d_in[3],  (const float*)d_in[4],
     (const float*)d_in[5],  (const float*)d_in[6],  (const float*)d_in[7],  (const float*)d_in[8]},
    {(const float*)d_in[9],  (const float*)d_in[10], (const float*)d_in[11],
     (const float*)d_in[12], (const float*)d_in[13], (const float*)d_in[14], (const float*)d_in[15]}
  };

  char* wp = (char*)d_ws;
  auto alloc = [&](size_t bytes) -> char* {
    char* r = wp;
    wp += (bytes + 255) & ~(size_t)255;
    return r;
  };
  float* h     = (float*)alloc(sizeof(float) * 33554432ULL);        // [B][256][4096]
  float* temp  = (float*)alloc(sizeof(float) * 6ULL * OPST);        // [6][B][64][4096]
  bf16*  xbf   = (bf16*)alloc(2ULL * 8388608ULL);                   // [B][4096][64] bf16
  bf16*  wb1   = (bf16*)alloc(2ULL * 4096);
  bf16*  wb3   = (bf16*)alloc(2ULL * 36864);
  bf16*  wb5   = (bf16*)alloc(2ULL * 102400);
  float* meanb = (float*)alloc(4ULL * 8192);
  float* maxb  = (float*)alloc(4ULL * 8192);
  float* nl    = (float*)alloc(4ULL * 8192);
  int*   idx   = (int*)alloc(4ULL * 64);
  float* mv    = (float*)alloc(4ULL * 256);
  float* tmean = (float*)alloc(4ULL * 12288);
  float* yb    = (float*)alloc(4ULL * 12288);

  for (int s = 0; s < 2; ++s) {
    const float* src = (s == 0) ? x : h;
    k_relu_stats<<<dim3(CFULL, BATCH), 256, 0, stream>>>(src, h, meanb, maxb);
    k_ca_mlp<<<BATCH, 256, 0, stream>>>(meanb, maxb, st[s].caw1, st[s].caw2, nl);
    k_topk<<<1, 256, 0, stream>>>(nl, idx);
    k_scale<<<dim3(CFULL, BATCH), 256, 0, stream>>>(h, nl);
    float* t5 = temp + 5ULL * OPST;
    k_gather<<<dim3(CH, BATCH), 256, 0, stream>>>(h, idx, t5, xbf);
    k_wcvt<<<16, 256, 0, stream>>>(st[s].w1, wb1, 1);
    k_wcvt<<<144, 256, 0, stream>>>(st[s].w3, wb3, 9);
    k_wcvt<<<400, 256, 0, stream>>>(st[s].w5, wb5, 25);
    k_conv<1><<<dim3(128, BATCH), 256, 0, stream>>>(xbf, wb1, temp);
    k_conv<3><<<dim3(128, BATCH), 256, 0, stream>>>(xbf, wb3, temp + OPST);
    k_conv<5><<<dim3(128, BATCH), 256, 0, stream>>>(xbf, wb5, temp + 2ULL * OPST);
    float* t3 = temp + 3ULL * OPST;
    float* t4 = temp + 4ULL * OPST;
    k_pool<<<dim3(CH, BATCH), 256, 0, stream>>>(t5, t3, t4);
    k_bnstats<<<dim3(CH, 2), 256, 0, stream>>>(t3, t4, mv);
    k_bnapply<<<dim3(CH, BATCH, 2), 256, 0, stream>>>(t3, t4, mv);
    k_tmean<<<dim3(AC, BATCH), 256, 0, stream>>>(temp, tmean);
    k_att<<<BATCH, 256, 0, stream>>>(tmean, st[s].aw1, st[s].aw2, yb);
    k_combine<<<dim3(CH, BATCH), 256, 0, stream>>>(temp, yb, idx, h);
  }
  k_add<<<8192, 256, 0, stream>>>(h, x, (float*)d_out, 33554432ULL);
}
